// DownEdgeMP_69415261438105
// MI455X (gfx1250) — matrix operand order repaired and checked by reading, not compiled, not run on hardware
//
#include <hip/hip_runtime.h>
#include <math.h>

// ---------------------------------------------------------------------------
// Types for CDNA5 WMMA (wave32): v_wmma_f32_16x16x32_bf16
// ---------------------------------------------------------------------------
typedef __attribute__((ext_vector_type(16))) __bf16 v16bf;
typedef __attribute__((ext_vector_type(8)))  float  v8f;

struct __attribute__((aligned(16))) U4 { unsigned int x, y, z, w; };
struct __attribute__((aligned(16))) F4 { float x, y, z, w; };
struct __attribute__((aligned(8)))  U2 { unsigned int x, y; };

union FragBF {
    v16bf v;
    unsigned int u[8];
};

__device__ __forceinline__ unsigned short f2bf(float f) {
    unsigned int u = __float_as_uint(f);
    u += 0x7FFFu + ((u >> 16) & 1u);          // round-to-nearest-even
    return (unsigned short)(u >> 16);
}

// pack 4 floats -> 4 bf16 (8 bytes) and store (8B-aligned, lowers to ds_store_b64)
__device__ __forceinline__ void store_bf4(unsigned short* dst, F4 v) {
    U2 o;
    o.x = (unsigned int)f2bf(v.x) | ((unsigned int)f2bf(v.y) << 16);
    o.y = (unsigned int)f2bf(v.z) | ((unsigned int)f2bf(v.w) << 16);
    *(U2*)dst = o;
}

__device__ __forceinline__ float selu_f(float x) {
    const float sc = 1.0507009873554805f;
    const float al = 1.6732632423543772f;
    return x > 0.f ? sc * x : sc * al * (__expf(x) - 1.f);
}

// ---------------------------------------------------------------------------
// Fragment loaders.
//
// A (16x32 bf16, from LDS tile, row stride STRIDE elements):
//   lanes 0-15 : row M = lane,     K = 0..15 + 32*kc
//   lanes 16-31: row M = lane-16,  K = 16..31 + 32*kc
//
// B (32x16 bf16, from global Wt stored [N=128, Kpad] row-major):
//   lanes 0-15 : col N = lane,     K = kc*32 + 0..15 (contiguous per lane)
//   lanes 16-31: col N = lane-16,  K = kc*32 + 16..31
// ---------------------------------------------------------------------------
__device__ __forceinline__ v16bf load_a_frag(const unsigned short* p /* row base + kc*32 */,
                                             int lane) {
    p += (lane >= 16) ? 16 : 0;
    FragBF f;
#pragma unroll
    for (int j = 0; j < 4; ++j)
        f.u[j] = *(const unsigned int*)(p + 2 * j);
#pragma unroll
    for (int j = 0; j < 4; ++j)
        f.u[4 + j] = *(const unsigned int*)(p + 8 + 2 * j);
    return f.v;
}

__device__ __forceinline__ v16bf load_b_frag(const unsigned short* p /* Wt + n*Kpad + kc*32 */,
                                             int lane) {
    p += (lane >= 16) ? 16 : 0;
    U4 lo = *(const U4*)(p);
    U4 hi = *(const U4*)(p + 8);
    FragBF f;
    f.u[0] = lo.x; f.u[1] = lo.y; f.u[2] = lo.z; f.u[3] = lo.w;
    f.u[4] = hi.x; f.u[5] = hi.y; f.u[6] = hi.z; f.u[7] = hi.w;
    return f.v;
}

// One 16-row M-tile x all 8 N-tiles (128 output cols), K = kChunks*32.
template <int STRIDE>
__device__ __forceinline__ void mlp_tile(const unsigned short* sX,
                                         const unsigned short* __restrict__ Wt,
                                         int kPad, int kChunks,
                                         int wave, int lane, v8f acc[8]) {
    v8f z = {};
#pragma unroll
    for (int n = 0; n < 8; ++n) acc[n] = z;
    const int mRow = wave * 16 + (lane & 15);
    const int nCol = lane & 15;
    for (int kc = 0; kc < kChunks; ++kc) {
        v16bf a = load_a_frag(sX + mRow * STRIDE + kc * 32, lane);
#pragma unroll
        for (int n = 0; n < 8; ++n) {
            v16bf b = load_b_frag(Wt + (size_t)(n * 16 + nCol) * kPad + kc * 32, lane);
            acc[n] = __builtin_amdgcn_wmma_f32_16x16x32_bf16(
                false, a, false, b, (short)0, acc[n], false, false);
        }
    }
}

// bias + SELU, write bf16 back into LDS tile (next layer's A input).
// C/D layout: VGPR r, lanes 0-15 -> M=r, lanes 16-31 -> M=r+8; N = lane&15.
template <int STRIDE>
__device__ __forceinline__ void epi_selu(unsigned short* sX, v8f acc[8],
                                         const float* __restrict__ bias,
                                         int wave, int lane) {
    const int mhalf = (lane >= 16) ? 8 : 0;
    const int nloc  = lane & 15;
#pragma unroll
    for (int nt = 0; nt < 8; ++nt) {
        float b = bias[nt * 16 + nloc];
#pragma unroll
        for (int r = 0; r < 8; ++r) {
            int m = wave * 16 + mhalf + r;
            sX[m * STRIDE + nt * 16 + nloc] = f2bf(selu_f(acc[nt][r] + b));
        }
    }
}

// ---------------------------------------------------------------------------
// Prep kernels
// ---------------------------------------------------------------------------
__global__ void transpose_w_kernel(const float* __restrict__ W,     // [K, 128]
                                   unsigned short* __restrict__ Wt, // [128, Kpad]
                                   int K, int Kpad) {
    int i = blockIdx.x * blockDim.x + threadIdx.x;
    if (i >= 128 * Kpad) return;
    int n = i / Kpad;
    int k = i - n * Kpad;
    float v = (k < K) ? W[(size_t)k * 128 + n] : 0.f;
    Wt[i] = f2bf(v);
}

__global__ void zero_f32_kernel(float* __restrict__ p, int n) {
    for (int i = blockIdx.x * blockDim.x + threadIdx.x; i < n;
         i += gridDim.x * blockDim.x)
        p[i] = 0.f;
}

// ---------------------------------------------------------------------------
// Stage 1: edge MLP (gather -> 3-layer MLP -> atomic scatter-add)
// Block = 256 threads (8 wave32), tile = 128 edges.
// ---------------------------------------------------------------------------
__global__ __launch_bounds__(256)
void edge_mlp_kernel(const float* __restrict__ e1, const float* __restrict__ e2,
                     const float* __restrict__ a12, const int* __restrict__ aidx,
                     const unsigned short* __restrict__ W0t,  // [128,288]
                     const unsigned short* __restrict__ W1t,  // [128,128]
                     const unsigned short* __restrict__ W2t,  // [128,128]
                     const float* __restrict__ b0, const float* __restrict__ b1,
                     const float* __restrict__ b2,
                     float* __restrict__ sums, float* __restrict__ cnt, int A) {
    __shared__ unsigned short sX[128 * 288];   // 73,728 B (LDS; WGP has 320 KB)
    const int tid  = threadIdx.x;
    const int lane = tid & 31;
    const int wave = tid >> 5;
    const int base = blockIdx.x * 128;
    const int* rowI = aidx;
    const int* colI = aidx + A;

    // ---- stage layer-0 input: [128 x 288] bf16 = [a12 | e1[row] | e2[col] | 0pad]
    // Vectorized: float4 gathers (global_load_b128) + packed bf16 (ds_store_b64).
    // Segment boundaries 8 / 136 / 264 are all multiples of 4.
    {
        int m = tid >> 1;
        int half = tid & 1;
        int e = base + m; if (e >= A) e = A - 1;
        int r = rowI[e], c = colI[e];
        const float* pa = a12 + (size_t)e * 8;
        const float* p1 = e1 + (size_t)r * 128;
        const float* p2 = e2 + (size_t)c * 128;
        unsigned short* dst = sX + m * 288;
        F4 zero4 = {0.f, 0.f, 0.f, 0.f};
        for (int g = half * 36; g < half * 36 + 36; ++g) {  // 36 float4 groups / half
            int j = g * 4;
            F4 v;
            if (j < 8)        v = *(const F4*)(pa + j);
            else if (j < 136) v = *(const F4*)(p1 + (j - 8));
            else if (j < 264) v = *(const F4*)(p2 + (j - 136));
            else              v = zero4;
            store_bf4(dst + j, v);
        }
    }
    __syncthreads();

    v8f acc[8];
    // layer 0: K=288 (264 padded), SELU
    mlp_tile<288>(sX, W0t, 288, 9, wave, lane, acc);
    epi_selu<288>(sX, acc, b0, wave, lane);
    __syncthreads();
    // layer 1: K=128, SELU
    mlp_tile<288>(sX, W1t, 128, 4, wave, lane, acc);
    epi_selu<288>(sX, acc, b1, wave, lane);
    __syncthreads();
    // layer 2: K=128, no activation -> scatter-add into sums[col]
    mlp_tile<288>(sX, W2t, 128, 4, wave, lane, acc);
    {
        const int mhalf = (lane >= 16) ? 8 : 0;
        const int nloc  = lane & 15;
        float bias[8];
#pragma unroll
        for (int nt = 0; nt < 8; ++nt) bias[nt] = b2[nt * 16 + nloc];
#pragma unroll
        for (int r = 0; r < 8; ++r) {
            int e = base + wave * 16 + mhalf + r;
            if (e < A) {
                size_t coff = (size_t)colI[e] * 128 + nloc;
#pragma unroll
                for (int nt = 0; nt < 8; ++nt)
                    atomicAdd(&sums[coff + nt * 16], acc[nt][r] + bias[nt]);
            }
        }
    }
    if (tid < 128) {
        int e = base + tid;
        if (e < A) atomicAdd(&cnt[colI[e]], 1.0f);
    }
}

// ---------------------------------------------------------------------------
// Stage 2: node MLP on concat(scatter-mean, e2) -> d_out
// ---------------------------------------------------------------------------
__global__ __launch_bounds__(256)
void node_mlp_kernel(const float* __restrict__ e2,
                     const float* __restrict__ sums, const float* __restrict__ cnt,
                     const unsigned short* __restrict__ W0t,  // [128,256]
                     const unsigned short* __restrict__ W1t,  // [128,128]
                     const unsigned short* __restrict__ W2t,  // [128,128]
                     const float* __restrict__ b0, const float* __restrict__ b1,
                     const float* __restrict__ b2,
                     float* __restrict__ out, int E2n) {
    __shared__ unsigned short sX[128 * 256];   // 65,536 B
    const int tid  = threadIdx.x;
    const int lane = tid & 31;
    const int wave = tid >> 5;
    const int base = blockIdx.x * 128;

    // ---- stage: [128 x 256] bf16 = [ sums/max(cnt,1) | e2 ]  (clamped rows)
    {
        int m = tid >> 1;
        int half = tid & 1;
        int node = base + m;
        int nc = node < E2n ? node : E2n - 1;
        unsigned short* dst = sX + m * 256;
        if (half == 0) {
            float inv = 1.f / fmaxf(cnt[nc], 1.f);
            const float* ps = sums + (size_t)nc * 128;
            for (int g = 0; g < 32; ++g) {              // 32 float4 groups
                F4 v = *(const F4*)(ps + g * 4);
                v.x *= inv; v.y *= inv; v.z *= inv; v.w *= inv;
                store_bf4(dst + g * 4, v);
            }
        } else {
            const float* pe = e2 + (size_t)nc * 128;
            for (int g = 0; g < 32; ++g)
                store_bf4(dst + 128 + g * 4, *(const F4*)(pe + g * 4));
        }
    }
    __syncthreads();

    v8f acc[8];
    mlp_tile<256>(sX, W0t, 256, 8, wave, lane, acc);
    epi_selu<256>(sX, acc, b0, wave, lane);
    __syncthreads();
    mlp_tile<256>(sX, W1t, 128, 4, wave, lane, acc);
    epi_selu<256>(sX, acc, b1, wave, lane);
    __syncthreads();
    mlp_tile<256>(sX, W2t, 128, 4, wave, lane, acc);
    {
        const int mhalf = (lane >= 16) ? 8 : 0;
        const int nloc  = lane & 15;
#pragma unroll
        for (int nt = 0; nt < 8; ++nt) {
            float b = b2[nt * 16 + nloc];
#pragma unroll
            for (int r = 0; r < 8; ++r) {
                int node = base + wave * 16 + mhalf + r;
                if (node < E2n)
                    out[(size_t)node * 128 + nt * 16 + nloc] = acc[nt][r] + b;
            }
        }
    }
}

// ---------------------------------------------------------------------------
// Host entry
// ---------------------------------------------------------------------------
extern "C" void kernel_launch(void* const* d_in, const int* in_sizes, int n_in,
                              void* d_out, int out_size, void* d_ws, size_t ws_size,
                              hipStream_t stream) {
    const float* e1  = (const float*)d_in[0];
    const float* e2  = (const float*)d_in[1];
    const float* a12 = (const float*)d_in[2];
    const int*  aidx = (const int*)d_in[3];
    const float* aW0 = (const float*)d_in[4];  const float* ab0 = (const float*)d_in[5];
    const float* aW1 = (const float*)d_in[6];  const float* ab1 = (const float*)d_in[7];
    const float* aW2 = (const float*)d_in[8];  const float* ab2 = (const float*)d_in[9];
    const float* eW0 = (const float*)d_in[10]; const float* eb0 = (const float*)d_in[11];
    const float* eW1 = (const float*)d_in[12]; const float* eb1 = (const float*)d_in[13];
    const float* eW2 = (const float*)d_in[14]; const float* eb2 = (const float*)d_in[15];
    float* out = (float*)d_out;

    const int E2n = in_sizes[1] / 128;      // 100000
    const int A   = in_sizes[2] / 8;        // 400000

    // workspace layout: [sums f32 E2*128][cnt f32 E2][bf16 weights...]
    char* ws = (char*)d_ws;
    float* sums = (float*)ws;
    float* cnt  = (float*)(ws + (size_t)E2n * 128 * 4);
    unsigned short* wb = (unsigned short*)(ws + (size_t)E2n * 128 * 4 + (size_t)E2n * 4);
    unsigned short* aW0t = wb;                  // 128*288
    unsigned short* aW1t = aW0t + 128 * 288;    // 128*128
    unsigned short* aW2t = aW1t + 128 * 128;    // 128*128
    unsigned short* eW0t = aW2t + 128 * 128;    // 128*256
    unsigned short* eW1t = eW0t + 128 * 256;    // 128*128
    unsigned short* eW2t = eW1t + 128 * 128;    // 128*128

    // prep: weight transpose/convert + zero accumulators
    transpose_w_kernel<<<(128 * 288 + 255) / 256, 256, 0, stream>>>(aW0, aW0t, 264, 288);
    transpose_w_kernel<<<(128 * 128 + 255) / 256, 256, 0, stream>>>(aW1, aW1t, 128, 128);
    transpose_w_kernel<<<(128 * 128 + 255) / 256, 256, 0, stream>>>(aW2, aW2t, 128, 128);
    transpose_w_kernel<<<(128 * 256 + 255) / 256, 256, 0, stream>>>(eW0, eW0t, 256, 256);
    transpose_w_kernel<<<(128 * 128 + 255) / 256, 256, 0, stream>>>(eW1, eW1t, 128, 128);
    transpose_w_kernel<<<(128 * 128 + 255) / 256, 256, 0, stream>>>(eW2, eW2t, 128, 128);
    zero_f32_kernel<<<2048, 256, 0, stream>>>(sums, E2n * 128 + E2n);

    // stage 1: edges
    edge_mlp_kernel<<<(A + 127) / 128, 256, 0, stream>>>(
        e1, e2, a12, aidx, aW0t, aW1t, aW2t, ab0, ab1, ab2, sums, cnt, A);

    // stage 2: nodes
    node_mlp_kernel<<<(E2n + 127) / 128, 256, 0, stream>>>(
        e2, sums, cnt, eW0t, eW1t, eW2t, eb0, eb1, eb2, out, E2n);
}
